// SSM_62294205661724
// MI455X (gfx1250) — matrix operand order repaired and checked by reading, not compiled, not run on hardware
//
#include <hip/hip_runtime.h>
#include <hip/hip_bf16.h>

// ---------------------------------------------------------------------------
// Hierarchical SSM forward loss on gfx1250 (MI455X).
// All conv / deconv / FC layers are lowered to one implicit-GEMM kernel built
// on v_wmma_f32_16x16x32_bf16 (wave32, 8 waves/block, 128x64x32 tiles,
// double-buffered LDS, register-staged tiles, shift-only addressing,
// clamp-instead-of-predicate bounds handling, hardware bf16 conversion).
// Prior recurrence is the only sequential part; posterior/decoder/losses are
// batched over all T*B rows.
// ---------------------------------------------------------------------------

typedef __bf16 bf16x2  __attribute__((ext_vector_type(2)));
typedef __bf16 bf16x8  __attribute__((ext_vector_type(8)));
typedef __bf16 bf16x16 __attribute__((ext_vector_type(16)));
typedef float  f32x8   __attribute__((ext_vector_type(8)));

#define BM 128
#define BN 64
#define BK 32
#define LDT 40   // padded LDS row stride (shorts): 80B -> conflict-free, 16B-aligned chunks

// hardware conversion: lowers to v_cvt_pk_bf16_f32 (RNE)
__device__ __forceinline__ unsigned pk2(float x, float y) {
    bf16x2 v;
    v.x = (__bf16)x;
    v.y = (__bf16)y;
    return __builtin_bit_cast(unsigned, v);
}
__device__ __forceinline__ unsigned short bf1(float x) {
    __bf16 h = (__bf16)x;
    return __builtin_bit_cast(unsigned short, h);
}

// AMODE: 0 = dense row-major [M,K]
//        1 = conv gather   (stride-2 4x4 conv, pad 1, NCHW input, k=(ci*16+ky*4+kx), H=2*OH)
//        2 = deconv gather (stride-2 4x4 conv_transpose, zero-filled taps, H=OH/2)
// BT   : B layout 0=[K,N] row-major, 1=[N,K] row-major (conv OIHW weights)
// CMODE: 0 = dense [M,N] store, 1 = NCHW scatter (img = m >> 2*oh_sh)
template<int AMODE, int BT, int CMODE>
__global__ __launch_bounds__(256)
void gemm_wmma_k(int M, int N, int K,
                 const float* __restrict__ A,
                 const float* __restrict__ Bw,
                 const float* __restrict__ bias,
                 float* __restrict__ C,
                 int Cin, int oh_sh, int act)
{
    constexpr int ASZ = BM * LDT, BSZ = BN * LDT;
    __shared__ unsigned short As[2 * ASZ];
    __shared__ unsigned short Bs[2 * BSZ];

    const int tid  = threadIdx.x;
    const int lane = tid & 31;
    const int wid  = tid >> 5;
    const int wm   = wid & 3;     // wave position along M (4)
    const int wn   = wid >> 2;    // wave position along N (2)
    const int m0   = blockIdx.y * BM;
    const int n0   = blockIdx.x * BN;
    const int l16  = lane & 15;
    const int half = lane >> 4;

    f32x8 acc[2][2];
    for (int i = 0; i < 2; ++i)
        for (int j = 0; j < 2; ++j)
            for (int e = 0; e < 8; ++e) acc[i][j][e] = 0.f;

    float a_reg[16];
    float b_reg[8];

    // ---- global -> register staging (implicit im2col, shift-only addressing) ----
    // Bounds policy: row (m) / col (n) overflow is CLAMPED (those C elements are
    // never stored, so garbage is harmless and we avoid exec-mask churn).
    // k-overflow is zeroed on the A side only (uniform branch, K%32!=0 is rare);
    // B's k-overflow garbage then multiplies A zeros.
    auto loadA = [&](int k0) {
        if (AMODE == 0) {
            const bool kfull = (k0 + BK <= K);          // uniform
#pragma unroll
            for (int i = 0; i < 4; ++i) {               // 1024 quads / 256 thr
                int idx = tid + i * 256;
                int row = idx >> 3, cg = (idx & 7) << 2;
                int m = m0 + row, k = k0 + cg;
                int mc = m < M ? m : (M - 1);
                float4 v;
                if (kfull) {
                    v = *(const float4*)&A[(size_t)mc * K + k];
                } else {
                    v = make_float4(0.f, 0.f, 0.f, 0.f);
                    if (k < K) v = *(const float4*)&A[(size_t)mc * K + k];
                }
                a_reg[4*i+0] = v.x; a_reg[4*i+1] = v.y;
                a_reg[4*i+2] = v.z; a_reg[4*i+3] = v.w;
            }
        } else {
            const int OH = 1 << oh_sh;
            const int H  = (AMODE == 1) ? (OH << 1) : (OH >> 1);
#pragma unroll
            for (int i = 0; i < 8; ++i) {               // 2048 pairs / 256 thr
                int pidx = tid + i * 256;
                int row = pidx >> 4, cp = (pidx & 15) << 1;
                int m  = m0 + row;
                int mc = m < M ? m : (M - 1);
                int img = mc >> (2 * oh_sh);
                int p   = mc & ((1 << (2 * oh_sh)) - 1);
                int oy  = p >> oh_sh, ox = p & (OH - 1);
#pragma unroll
                for (int j = 0; j < 2; ++j) {
                    int k = k0 + cp + j;
                    float v = 0.f;
                    if (k < K) {                        // address safety (ci < Cin)
                        int ci = k >> 4, r = k & 15, ky = r >> 2, kx = r & 3;
                        if (AMODE == 1) {               // conv: iy = 2*oy-1+ky
                            int iy = oy * 2 - 1 + ky, ix = ox * 2 - 1 + kx;
                            if (iy >= 0 && iy < H && ix >= 0 && ix < H)
                                v = A[(((size_t)img * Cin + ci) * H + iy) * H + ix];
                        } else {                        // deconv: iy = (oy-1+ky)/2 if even
                            int iyn = oy - 1 + ky, ixn = ox - 1 + kx;
                            if (iyn >= 0 && ixn >= 0 && !(iyn & 1) && !(ixn & 1)) {
                                int iy = iyn >> 1, ix = ixn >> 1;
                                if (iy < H && ix < H)
                                    v = A[(((size_t)img * Cin + ci) * H + iy) * H + ix];
                            }
                        }
                    }
                    a_reg[2*i + j] = v;
                }
            }
        }
    };
    auto loadB = [&](int k0) {
        if (BT) {
#pragma unroll
            for (int i = 0; i < 2; ++i) {               // 512 quads / 256 thr
                int idx = tid + i * 256;
                int row = idx >> 3, cg = (idx & 7) << 2;
                int n = n0 + row, k = k0 + cg;
                int nc = n < N ? n : (N - 1);
                int kc = k + 3 < K ? k : 0;             // k-overflow -> clamp (A side is 0)
                float4 v = *(const float4*)&Bw[(size_t)nc * K + kc];
                b_reg[4*i+0] = v.x; b_reg[4*i+1] = v.y;
                b_reg[4*i+2] = v.z; b_reg[4*i+3] = v.w;
            }
        } else {
#pragma unroll
            for (int i = 0; i < 2; ++i) {               // one k, 4 consecutive n
                int idx = tid + i * 256;
                int krow = idx >> 4, ng = (idx & 15) << 2;
                int k = k0 + krow, n = n0 + ng;
                int kc = k < K ? k : (K - 1);           // clamp (A side is 0 past K)
                int nc = n + 3 < N ? n : (N - 4);       // N is a multiple of 4 here
                float4 v = *(const float4*)&Bw[(size_t)kc * N + nc];
                b_reg[4*i+0] = v.x; b_reg[4*i+1] = v.y;
                b_reg[4*i+2] = v.z; b_reg[4*i+3] = v.w;
            }
        }
    };
    // ---- register -> LDS staging (hardware f32->bf16, packed stores) ----
    auto storeA = [&](unsigned short* dst) {
        if (AMODE == 0) {
#pragma unroll
            for (int i = 0; i < 4; ++i) {
                int idx = tid + i * 256;
                int row = idx >> 3, cg = (idx & 7) << 2;
                uint2 pk;
                pk.x = pk2(a_reg[4*i+0], a_reg[4*i+1]);
                pk.y = pk2(a_reg[4*i+2], a_reg[4*i+3]);
                *reinterpret_cast<uint2*>(&dst[row * LDT + cg]) = pk;
            }
        } else {
#pragma unroll
            for (int i = 0; i < 8; ++i) {
                int pidx = tid + i * 256;
                int row = pidx >> 4, cp = (pidx & 15) << 1;
                *reinterpret_cast<unsigned*>(&dst[row * LDT + cp]) =
                    pk2(a_reg[2*i], a_reg[2*i+1]);
            }
        }
    };
    auto storeB = [&](unsigned short* dst) {
        if (BT) {
#pragma unroll
            for (int i = 0; i < 2; ++i) {
                int idx = tid + i * 256;
                int row = idx >> 3, cg = (idx & 7) << 2;
                uint2 pk;
                pk.x = pk2(b_reg[4*i+0], b_reg[4*i+1]);
                pk.y = pk2(b_reg[4*i+2], b_reg[4*i+3]);
                *reinterpret_cast<uint2*>(&dst[row * LDT + cg]) = pk;
            }
        } else {
#pragma unroll
            for (int i = 0; i < 2; ++i) {               // transpose to Bs[n][k]
                int idx = tid + i * 256;
                int krow = idx >> 4, ng = (idx & 15) << 2;
                dst[(ng + 0) * LDT + krow] = bf1(b_reg[4*i+0]);
                dst[(ng + 1) * LDT + krow] = bf1(b_reg[4*i+1]);
                dst[(ng + 2) * LDT + krow] = bf1(b_reg[4*i+2]);
                dst[(ng + 3) * LDT + krow] = bf1(b_reg[4*i+3]);
            }
        }
    };

    const int kSteps = (K + BK - 1) / BK;
    loadA(0); loadB(0);
    storeA(As); storeB(Bs);
    __syncthreads();

    for (int kt = 0; kt < kSteps; ++kt) {
        const unsigned short* Ac = As + (kt & 1) * ASZ;
        const unsigned short* Bc = Bs + (kt & 1) * BSZ;

        // 1) fragment ds_loads from current buffer
        //    A 16-bit 16x32: lane half -> K base {0,8}; chunks [b,b+8) and [b+16,b+24)
        //    B 16-bit 32x16: lane half -> K base {0,8}; chunks [b,b+8) and [b+16,b+24)
        bf16x16 af[2], bfr[2];
#pragma unroll
        for (int i = 0; i < 2; ++i) {
            int row = wm * 32 + i * 16 + l16;
            int ka  = half * 8;
            bf16x8 lo = *reinterpret_cast<const bf16x8*>(&Ac[row * LDT + ka]);
            bf16x8 hi = *reinterpret_cast<const bf16x8*>(&Ac[row * LDT + ka + 16]);
            af[i] = __builtin_shufflevector(lo, hi, 0,1,2,3,4,5,6,7,8,9,10,11,12,13,14,15);
        }
#pragma unroll
        for (int j = 0; j < 2; ++j) {
            int row = wn * 32 + j * 16 + l16;
            int kb  = half * 8;
            bf16x8 lo = *reinterpret_cast<const bf16x8*>(&Bc[row * LDT + kb]);
            bf16x8 hi = *reinterpret_cast<const bf16x8*>(&Bc[row * LDT + kb + 16]);
            bfr[j] = __builtin_shufflevector(lo, hi, 0,1,2,3,4,5,6,7,8,9,10,11,12,13,14,15);
        }

        // 2) issue next tile's global loads (latency overlaps the WMMAs below)
        if (kt + 1 < kSteps) { loadA((kt + 1) * BK); loadB((kt + 1) * BK); }

        // 3) matrix ops
#pragma unroll
        for (int i = 0; i < 2; ++i)
#pragma unroll
            for (int j = 0; j < 2; ++j)
                acc[i][j] = __builtin_amdgcn_wmma_f32_16x16x32_bf16(
                    false, af[i], false, bfr[j], (short)0, acc[i][j], false, false);

        // 4) write next tile into the other LDS buffer, single barrier per step
        if (kt + 1 < kSteps) {
            unsigned short* An = As + ((kt + 1) & 1) * ASZ;
            unsigned short* Bn = Bs + ((kt + 1) & 1) * BSZ;
            storeA(An); storeB(Bn);
        }
        __syncthreads();
    }

    // ---- epilogue: bias + act + (dense | NCHW) store ----
#pragma unroll
    for (int i = 0; i < 2; ++i)
#pragma unroll
        for (int j = 0; j < 2; ++j) {
            int gcol = n0 + wn * 32 + j * 16 + l16;
            if (gcol >= N) continue;
            float bv = bias ? bias[gcol] : 0.f;
#pragma unroll
            for (int e = 0; e < 8; ++e) {
                int grow = m0 + wm * 32 + i * 16 + half * 8 + e;
                if (grow >= M) continue;
                float v = acc[i][j][e] + bv;
                if (act) v = v > 0.f ? v : 0.f;
                if (CMODE == 0) {
                    C[(size_t)grow * N + gcol] = v;
                } else {
                    int img = grow >> (2 * oh_sh);
                    int p   = grow & ((1 << (2 * oh_sh)) - 1);
                    C[(((size_t)img * N + gcol) << (2 * oh_sh)) + p] = v;
                }
            }
        }
}

// ---------------------------------------------------------------------------
// Elementwise / reduction helpers
// ---------------------------------------------------------------------------
__global__ void zero_k(float* p, int n) {
    int i = blockIdx.x * blockDim.x + threadIdx.x;
    if (i < n) p[i] = 0.f;
}

__global__ void copy_cols_k(float* dst, int dld, int doff,
                            const float* src, int sld, int rows, int cols) {
    int i = blockIdx.x * blockDim.x + threadIdx.x;
    if (i >= rows * cols) return;
    int r = i / cols, c = i - r * cols;
    dst[(size_t)r * dld + doff + c] = src[(size_t)r * sld + c];
}

__global__ void softplus_k(float* p, int n) {
    int i = blockIdx.x * blockDim.x + threadIdx.x;
    if (i >= n) return;
    float v = p[i];
    p[i] = (v > 20.f ? v : log1pf(expf(v))) + 1e-5f;
}

__global__ void kl_reduce_k(const float* __restrict__ mq, const float* __restrict__ sq,
                            const float* __restrict__ mp, const float* __restrict__ sp,
                            int n, float scale, float* out) {
    __shared__ float sm[256];
    float acc = 0.f;
    for (int i = blockIdx.x * blockDim.x + threadIdx.x; i < n; i += gridDim.x * blockDim.x) {
        float a = mq[i], b = sq[i], c = mp[i], d = sp[i];
        float dm = a - c;
        acc += logf(d / b) + (b * b + dm * dm) / (2.f * d * d) - 0.5f;
    }
    sm[threadIdx.x] = acc; __syncthreads();
    for (int s = 128; s > 0; s >>= 1) {
        if (threadIdx.x < s) sm[threadIdx.x] += sm[threadIdx.x + s];
        __syncthreads();
    }
    if (threadIdx.x == 0) atomicAdd(out, sm[0] * scale);
}

__global__ void nll_reduce_k(const float* __restrict__ x, const float* __restrict__ loc,
                             int n, float scale, float* out) {
    __shared__ float sm[256];
    float acc = 0.f;
    for (int i = blockIdx.x * blockDim.x + threadIdx.x; i < n; i += gridDim.x * blockDim.x) {
        float d = x[i] - loc[i];
        acc += 0.5f * d * d + 0.91893853320467274f;   // 0.5*log(2*pi)
    }
    sm[threadIdx.x] = acc; __syncthreads();
    for (int s = 128; s > 0; s >>= 1) {
        if (threadIdx.x < s) sm[threadIdx.x] += sm[threadIdx.x + s];
        __syncthreads();
    }
    if (threadIdx.x == 0) atomicAdd(out, sm[0] * scale);
}

// ---------------------------------------------------------------------------
// Host orchestration
// ---------------------------------------------------------------------------
namespace {

constexpr int T_ = 30, B_ = 32, S_ = 64, A_ = 8, HID_ = 256, HD_ = 1024;
constexpr int CHUNK = 480;   // images per encoder/decoder chunk (960 = 2*480)

struct Params {
    const float *enc_w[2][4], *enc_b[2][4], *enc_wf[2], *enc_bf[2];
    const float *dec_w[2][4], *dec_b[2][4], *dec_wf[2], *dec_bf[2];
    const float *pri_w1[2], *pri_b1[2], *pri_wl[2], *pri_bl[2], *pri_ws[2], *pri_bs[2];
    const float *post_w1[2], *post_b1[2], *post_wl[2], *post_bl[2], *post_ws[2], *post_bs[2];
};

// amode: 0 dense (BT=0, dense store), 1 conv, 2 deconv (BT=1, NCHW store)
inline void gemm(hipStream_t s, int M, int N, int K,
                 const float* A, const float* Bw, const float* bias, float* C,
                 int amode, int Cin, int oh_sh, int act) {
    dim3 g((N + BN - 1) / BN, (M + BM - 1) / BM);
    if (amode == 0)
        gemm_wmma_k<0, 0, 0><<<g, 256, 0, s>>>(M, N, K, A, Bw, bias, C, Cin, oh_sh, act);
    else if (amode == 1)
        gemm_wmma_k<1, 1, 1><<<g, 256, 0, s>>>(M, N, K, A, Bw, bias, C, Cin, oh_sh, act);
    else
        gemm_wmma_k<2, 1, 1><<<g, 256, 0, s>>>(M, N, K, A, Bw, bias, C, Cin, oh_sh, act);
}

inline void cpcols(hipStream_t s, float* dst, int dld, int doff,
                   const float* src, int sld, int rows, int cols) {
    int n = rows * cols;
    copy_cols_k<<<(n + 255) / 256, 256, 0, s>>>(dst, dld, doff, src, sld, rows, cols);
}

inline int rgrid(int n) { int g = (n + 255) / 256; return g > 4096 ? 4096 : g; }

} // namespace

extern "C" void kernel_launch(void* const* d_in, const int* in_sizes, int n_in,
                              void* d_out, int out_size, void* d_ws, size_t ws_size,
                              hipStream_t stream) {
    const float* x0 = (const float*)d_in[0];   // [32,3,64,64]
    const float* x  = (const float*)d_in[1];   // [30,32,3,64,64]
    const float* a  = (const float*)d_in[2];   // [30,32,8]
    auto F = [&](int i) { return (const float*)d_in[i]; };

    // ---- resolve param leaf ordering (jax-sorted vs dict-insertion) ----
    Params P;
    bool sorted = (n_in > 3 && in_sizes[3] == 128);   // sorted: dec0.b0(128); insertion: enc0.w0(1536)
    if (sorted) {
        for (int i = 0; i < 2; ++i) {
            int d = 3 + 10 * i;                        // dec_i: b0..b3,bf,w0..w3,wf
            for (int j = 0; j < 4; ++j) P.dec_b[i][j] = F(d + j);
            P.dec_bf[i] = F(d + 4);
            for (int j = 0; j < 4; ++j) P.dec_w[i][j] = F(d + 5 + j);
            P.dec_wf[i] = F(d + 9);
            int e = 23 + 10 * i;                       // enc_i: same key order
            for (int j = 0; j < 4; ++j) P.enc_b[i][j] = F(e + j);
            P.enc_bf[i] = F(e + 4);
            for (int j = 0; j < 4; ++j) P.enc_w[i][j] = F(e + 5 + j);
            P.enc_wf[i] = F(e + 9);
            int po = 43 + 6 * i;                       // post_i: b1,bl,bs,w1,wl,ws
            P.post_b1[i] = F(po); P.post_bl[i] = F(po + 1); P.post_bs[i] = F(po + 2);
            P.post_w1[i] = F(po + 3); P.post_wl[i] = F(po + 4); P.post_ws[i] = F(po + 5);
            int pr = 55 + 6 * i;                       // prior_i
            P.pri_b1[i] = F(pr); P.pri_bl[i] = F(pr + 1); P.pri_bs[i] = F(pr + 2);
            P.pri_w1[i] = F(pr + 3); P.pri_wl[i] = F(pr + 4); P.pri_ws[i] = F(pr + 5);
        }
    } else {
        for (int i = 0; i < 2; ++i) {
            int b = 3 + 32 * i;
            for (int j = 0; j < 4; ++j) { P.enc_w[i][j] = F(b + 2 * j); P.enc_b[i][j] = F(b + 2 * j + 1); }
            P.enc_wf[i] = F(b + 8); P.enc_bf[i] = F(b + 9);
            P.pri_w1[i] = F(b + 10); P.pri_b1[i] = F(b + 11);
            P.pri_wl[i] = F(b + 12); P.pri_bl[i] = F(b + 13);
            P.pri_ws[i] = F(b + 14); P.pri_bs[i] = F(b + 15);
            P.post_w1[i] = F(b + 16); P.post_b1[i] = F(b + 17);
            P.post_wl[i] = F(b + 18); P.post_bl[i] = F(b + 19);
            P.post_ws[i] = F(b + 20); P.post_bs[i] = F(b + 21);
            P.dec_wf[i] = F(b + 22); P.dec_bf[i] = F(b + 23);
            for (int j = 0; j < 4; ++j) { P.dec_w[i][j] = F(b + 24 + 2 * j); P.dec_b[i][j] = F(b + 24 + 2 * j + 1); }
        }
    }

    // ---- workspace arena (~110 MB) ----
    size_t off = 0;
    auto alloc = [&](size_t elems) {
        float* p = (float*)((char*)d_ws + off);
        off += ((elems * sizeof(float) + 255) / 256) * 256;
        return p;
    };
    float* bufA    = alloc((size_t)CHUNK * 32 * 32 * 32);  // 15.7M
    float* bufB    = alloc((size_t)CHUNK * 64 * 16 * 16);  // 7.9M
    float* hx[2]   = { alloc((size_t)T_ * B_ * HD_), alloc((size_t)T_ * B_ * HD_) };
    float* h0[2]   = { alloc((size_t)B_ * HD_), alloc((size_t)B_ * HD_) };
    float* s_seq[2]= { alloc((size_t)(T_ + 1) * B_ * S_), alloc((size_t)(T_ + 1) * B_ * S_) };
    float* sp_pre[2]={ alloc((size_t)T_ * B_ * S_), alloc((size_t)T_ * B_ * S_) };
    float* post_in = alloc((size_t)T_ * B_ * 1160);
    float* post_h  = alloc((size_t)T_ * B_ * HID_);
    float* mq      = alloc((size_t)T_ * B_ * S_);
    float* sq      = alloc((size_t)T_ * B_ * S_);
    float* pri_in  = alloc((size_t)B_ * 136);
    float* pri_h   = alloc((size_t)B_ * HID_);
    if (off > ws_size) return;   // workspace too small: bail deterministically

    float* out = (float*)d_out;
    zero_k<<<1, 64, 0, stream>>>(out, out_size);

    // ---- encoder: conv chain as implicit-GEMM, batched over all frames ----
    auto encoder = [&](int lvl, const float* X, int nimg, float* outH) {
        gemm(stream, nimg * 32 * 32,  32,   48, X,    P.enc_w[lvl][0], P.enc_b[lvl][0], bufA, 1,   3, 5, 1);
        gemm(stream, nimg * 16 * 16,  64,  512, bufA, P.enc_w[lvl][1], P.enc_b[lvl][1], bufB, 1,  32, 4, 1);
        gemm(stream, nimg *  8 *  8, 128, 1024, bufB, P.enc_w[lvl][2], P.enc_b[lvl][2], bufA, 1,  64, 3, 1);
        gemm(stream, nimg *  4 *  4, 256, 2048, bufA, P.enc_w[lvl][3], P.enc_b[lvl][3], bufB, 1, 128, 2, 1);
        gemm(stream, nimg, HD_, 4096, bufB, P.enc_wf[lvl], P.enc_bf[lvl], outH, 0, 0, 0, 0);
    };
    for (int lvl = 0; lvl < 2; ++lvl) {
        for (int c = 0; c < 2; ++c)    // 960 frames in 2 chunks
            encoder(lvl, x + (size_t)c * CHUNK * 3 * 64 * 64, CHUNK, hx[lvl] + (size_t)c * CHUNK * HD_);
        encoder(lvl, x0, B_, h0[lvl]);
    }

    // ---- s0: posterior mean with zeroed s_prev and actions ----
    for (int lvl = 0; lvl < 2; ++lvl) {
        int Kp = S_ + HD_ + A_ + (lvl ? S_ : 0);
        zero_k<<<(B_ * Kp + 255) / 256, 256, 0, stream>>>(post_in, B_ * Kp);
        cpcols(stream, post_in, Kp, S_, h0[lvl], HD_, B_, HD_);
        gemm(stream, B_, HID_, Kp, post_in, P.post_w1[lvl], P.post_b1[lvl], pri_h, 0, 0, 0, 1);
        gemm(stream, B_, S_, HID_, pri_h, P.post_wl[lvl], P.post_bl[lvl], s_seq[lvl], 0, 0, 0, 0);
    }

    // ---- prior recurrence (only sequential part; tiny MLPs) ----
    // s_seq[lvl] slot t+1 = prior mean at step t  (also the state fed forward)
    for (int t = 0; t < T_; ++t) {
        const float* a_t = a + (size_t)t * B_ * A_;
        for (int lvl = 0; lvl < 2; ++lvl) {
            int Kp = S_ + A_ + (lvl ? S_ : 0);
            cpcols(stream, pri_in, Kp, 0,       s_seq[lvl] + (size_t)t * B_ * S_, S_, B_, S_);
            cpcols(stream, pri_in, Kp, S_,      a_t, A_, B_, A_);
            if (lvl)
                cpcols(stream, pri_in, Kp, S_ + A_, s_seq[0] + (size_t)(t + 1) * B_ * S_, S_, B_, S_);
            gemm(stream, B_, HID_, Kp, pri_in, P.pri_w1[lvl], P.pri_b1[lvl], pri_h, 0, 0, 0, 1);
            gemm(stream, B_, S_, HID_, pri_h, P.pri_wl[lvl], P.pri_bl[lvl],
                 s_seq[lvl] + (size_t)(t + 1) * B_ * S_, 0, 0, 0, 0);
            gemm(stream, B_, S_, HID_, pri_h, P.pri_ws[lvl], P.pri_bs[lvl],
                 sp_pre[lvl] + (size_t)t * B_ * S_, 0, 0, 0, 0);
        }
    }

    // ---- posterior + KL, batched over all T*B rows ----
    const int TB = T_ * B_, nS = TB * S_;
    for (int lvl = 0; lvl < 2; ++lvl) {
        int Kp = S_ + HD_ + A_ + (lvl ? S_ : 0);
        cpcols(stream, post_in, Kp, 0,          s_seq[lvl],        S_,  TB, S_);   // s_prev (slots 0..T-1)
        cpcols(stream, post_in, Kp, S_,         hx[lvl],           HD_, TB, HD_);
        cpcols(stream, post_in, Kp, S_ + HD_,   a,                 A_,  TB, A_);
        if (lvl)
            cpcols(stream, post_in, Kp, S_ + HD_ + A_, s_seq[0] + B_ * S_, S_, TB, S_);  // s_t level0
        gemm(stream, TB, HID_, Kp, post_in, P.post_w1[lvl], P.post_b1[lvl], post_h, 0, 0, 0, 1);
        gemm(stream, TB, S_, HID_, post_h, P.post_wl[lvl], P.post_bl[lvl], mq, 0, 0, 0, 0);
        gemm(stream, TB, S_, HID_, post_h, P.post_ws[lvl], P.post_bs[lvl], sq, 0, 0, 0, 0);
        softplus_k<<<(nS + 255) / 256, 256, 0, stream>>>(sq, nS);
        softplus_k<<<(nS + 255) / 256, 256, 0, stream>>>(sp_pre[lvl], nS);
        kl_reduce_k<<<rgrid(nS), 256, 0, stream>>>(mq, sq, s_seq[lvl] + B_ * S_, sp_pre[lvl], nS,
                                                   1.0f / (float)B_, out);
    }

    // ---- decoder + NLL, batched over all T*B rows (chunked) ----
    for (int lvl = 0; lvl < 2; ++lvl) {
        for (int c = 0; c < 2; ++c) {
            int nimg = CHUNK;
            const float* Srows = s_seq[lvl] + B_ * S_ + (size_t)c * CHUNK * S_;
            const float* Xrows = x + (size_t)c * CHUNK * 3 * 64 * 64;
            gemm(stream, nimg, 4096, S_, Srows, P.dec_wf[lvl], P.dec_bf[lvl], bufB, 0, 0, 0, 1);
            gemm(stream, nimg *  8 *  8, 128, 4096, bufB, P.dec_w[lvl][0], P.dec_b[lvl][0], bufA, 2, 256, 3, 1);
            gemm(stream, nimg * 16 * 16,  64, 2048, bufA, P.dec_w[lvl][1], P.dec_b[lvl][1], bufB, 2, 128, 4, 1);
            gemm(stream, nimg * 32 * 32,  32, 1024, bufB, P.dec_w[lvl][2], P.dec_b[lvl][2], bufA, 2,  64, 5, 1);
            gemm(stream, nimg * 64 * 64,   3,  512, bufA, P.dec_w[lvl][3], P.dec_b[lvl][3], bufB, 2,  32, 6, 0);
            int n = nimg * 3 * 64 * 64;
            nll_reduce_k<<<rgrid(n), 256, 0, stream>>>(Xrows, bufB, n, 1.0f / (float)B_, out);
        }
    }
}
